// _TokenAttnLayer_43662637532066
// MI455X (gfx1250) — matrix operand order repaired and checked by reading, not compiled, not run on hardware
//
#include <hip/hip_runtime.h>

// ---------------------------------------------------------------------------
// TokenAttnLayer for MI455X (gfx1250): f16 WMMA pipeline
//   K1: Q/K/V projections (fp32 in -> f16 out, [bt,h,v,e] layout)
//   K2: per-(bt,head) attention with top-16 mask + softmax, all in LDS
//   K3: output projection + residual (fp32 out)
// ---------------------------------------------------------------------------

typedef __attribute__((ext_vector_type(16))) _Float16 v16h;
typedef __attribute__((ext_vector_type(8)))  float    v8f;

union F16x8 { uint4 u; _Float16 h[8]; };
union FragU { uint4 u[2]; v16h v; };

// A fragment (16x32 f16, M x K): lane m(0-15) holds K {0..7, 16..23},
// lane m+16 holds K {8..15, 24..31}  (per CDNA5 ISA 7.12.2)
__device__ __forceinline__ v16h load_a_frag(const _Float16* t, int mBase, int k0, int ld) {
  int lane = threadIdx.x & 31;
  int m = lane & 15, hi = lane >> 4;
  const _Float16* p = t + (size_t)(mBase + m) * ld + k0 + hi * 8;
  FragU f;
  f.u[0] = *(const uint4*)p;          // K +0..7
  f.u[1] = *(const uint4*)(p + 16);   // K +16..23
  return f.v;
}

// B fragment (32x16 f16, K x N) where source is row-major [N, K]:
// lane n(0-15) holds B[k0+{0..7,16..23}, n]; lanes 16-31: K {8..15, 24..31}
__device__ __forceinline__ v16h load_b_frag(const _Float16* t, int nBase, int k0, int ld) {
  int lane = threadIdx.x & 31;
  int n = lane & 15, hi = lane >> 4;
  const _Float16* p = t + (size_t)(nBase + n) * ld + k0 + hi * 8;
  FragU f;
  f.u[0] = *(const uint4*)p;
  f.u[1] = *(const uint4*)(p + 16);
  return f.v;
}

// C/D 16x16 f32 layout: VGPR i -> row mBase + i + 8*(lane>=16), col nBase + lane%16
__device__ __forceinline__ void store_c_f16(_Float16* t, v8f c, int mBase, int nBase,
                                            int ld, float scale) {
  int lane = threadIdx.x & 31;
  int n = lane & 15, hi = lane >> 4;
#pragma unroll
  for (int i = 0; i < 8; ++i)
    t[(mBase + hi * 8 + i) * ld + nBase + n] = (_Float16)(c[i] * scale);
}

__device__ __forceinline__ void store_c_f32(float* t, v8f c, int mBase, int nBase, int ld) {
  int lane = threadIdx.x & 31;
  int n = lane & 15, hi = lane >> 4;
#pragma unroll
  for (int i = 0; i < 8; ++i)
    t[(mBase + hi * 8 + i) * ld + nBase + n] = c[i];
}

#define WMMA_F16(a, b, c) \
  __builtin_amdgcn_wmma_f32_16x16x32_f16(false, (a), false, (b), (short)0, (c), false, false)

// ---------------------------------------------------------------------------
// Kernel 1: Q/K/V projections.  y = h @ W.T  (rows = b*V*T = 131072, d = 512)
// Output scattered to f16 [bt, h, v, e] with bt = b*64 + t.
// grid (1024, 8, 3), block 256 (8 waves)
// ---------------------------------------------------------------------------
__global__ __launch_bounds__(256)
void k1_qkv_proj(const float* __restrict__ h,
                 const float* __restrict__ Wq, const float* __restrict__ Wk,
                 const float* __restrict__ Wv,
                 _Float16* __restrict__ qb, _Float16* __restrict__ kb,
                 _Float16* __restrict__ vb) {
  __shared__ _Float16 sA[128 * 32];  // 8 KB  (A k-slice, f16)
  __shared__ _Float16 sB[64 * 32];   // 4 KB  (W k-slice, f16)
  __shared__ _Float16 sC[128 * 64];  // 16 KB (output staging)

  const int tid = threadIdx.x;
  const int wave = tid >> 5;
  const int rowBase = blockIdx.x * 128;
  const int hIdx = blockIdx.y;  // head index == 64-wide output column tile
  const int which = blockIdx.z;
  const float* W = (which == 0) ? Wq : (which == 1) ? Wk : Wv;
  _Float16* dst = (which == 0) ? qb : (which == 1) ? kb : vb;

  const v8f z8 = {0.f, 0.f, 0.f, 0.f, 0.f, 0.f, 0.f, 0.f};
  v8f acc[4] = {z8, z8, z8, z8};

  for (int kt = 0; kt < 16; ++kt) {
    __syncthreads();
    // A tile: 128x32 fp32 -> f16 LDS
#pragma unroll
    for (int i = 0; i < 4; ++i) {
      int q = tid + i * 256;               // 0..1023 float4s
      int r = q >> 3, c4 = q & 7;
      float4 a4 = *(const float4*)(h + (size_t)(rowBase + r) * 512 + kt * 32 + c4 * 4);
      _Float16* o = sA + r * 32 + c4 * 4;
      o[0] = (_Float16)a4.x; o[1] = (_Float16)a4.y;
      o[2] = (_Float16)a4.z; o[3] = (_Float16)a4.w;
    }
    // B tile: 64x32 fp32 -> f16 LDS (rows of W, row-major over K)
#pragma unroll
    for (int i = 0; i < 2; ++i) {
      int q = tid + i * 256;               // 0..511 float4s
      int r = q >> 3, c4 = q & 7;
      float4 b4 = *(const float4*)(W + (size_t)(hIdx * 64 + r) * 512 + kt * 32 + c4 * 4);
      _Float16* o = sB + r * 32 + c4 * 4;
      o[0] = (_Float16)b4.x; o[1] = (_Float16)b4.y;
      o[2] = (_Float16)b4.z; o[3] = (_Float16)b4.w;
    }
    __syncthreads();
    v16h a = load_a_frag(sA, wave * 16, 0, 32);
#pragma unroll
    for (int n = 0; n < 4; ++n) {
      v16h b = load_b_frag(sB, n * 16, 0, 32);
      acc[n] = WMMA_F16(a, b, acc[n]);
    }
  }
  __syncthreads();
#pragma unroll
  for (int n = 0; n < 4; ++n) store_c_f16(sC, acc[n], wave * 16, n * 16, 64, 1.0f);
  __syncthreads();

  // scatter: row g = (b*128+v)*64 + t  ->  dst[((bt*8+h)*128+v)*64 + e]
  {
    int r = tid >> 1, half = tid & 1;
    int g = rowBase + r;
    int b = g >> 13;
    int v = (g >> 6) & 127;
    int tt = g & 63;
    int bt = b * 64 + tt;
    uint4* d4 = (uint4*)(dst + (((size_t)bt * 8 + hIdx) * 128 + v) * 64 + half * 32);
    const uint4* s4 = (const uint4*)(sC + r * 64 + half * 32);
#pragma unroll
    for (int i = 0; i < 4; ++i) d4[i] = s4[i];
  }
}

// ---------------------------------------------------------------------------
// Kernel 2: attention per (bt, head). S = QK^T/8, top-16 mask, softmax, P@V.
// grid (1024, 8), block 256.  LDS = exactly 64 KB.
// ---------------------------------------------------------------------------
__global__ __launch_bounds__(256)
void k2_attention(const _Float16* __restrict__ qb, const _Float16* __restrict__ kb,
                  const _Float16* __restrict__ vb, _Float16* __restrict__ mixed) {
  __shared__ _Float16 sQV[128 * 64];   // 16 KB: Q, later V^T [e][v]
  __shared__ _Float16 sK[128 * 64];    // 16 KB: K, later mixed staging
  __shared__ _Float16 sSP[128 * 128];  // 32 KB: scores, then probs (in place)

  const int tid = threadIdx.x;
  const int wave = tid >> 5;
  const int bt = blockIdx.x;
  const int hIdx = blockIdx.y;
  const size_t base = ((size_t)bt * 8 + hIdx) * (128 * 64);
  const v8f z8 = {0.f, 0.f, 0.f, 0.f, 0.f, 0.f, 0.f, 0.f};

  // load Q, K tiles (contiguous f16)
  {
    const uint4* q4 = (const uint4*)(qb + base);
    const uint4* k4 = (const uint4*)(kb + base);
    uint4* sq = (uint4*)sQV;
    uint4* sk = (uint4*)sK;
#pragma unroll
    for (int i = 0; i < 4; ++i) {
      int idx = tid + i * 256;  // 1024 uint4s each
      sq[idx] = q4[idx];
      sk[idx] = k4[idx];
    }
  }
  __syncthreads();

  // S = Q K^T * (1/8):  M=128, N=128, K=64; wave w owns rows w*16..w*16+15
  {
    v8f acc[8] = {z8, z8, z8, z8, z8, z8, z8, z8};
#pragma unroll
    for (int kk = 0; kk < 64; kk += 32) {
      v16h a = load_a_frag(sQV, wave * 16, kk, 64);
#pragma unroll
      for (int n = 0; n < 8; ++n) {
        v16h b = load_b_frag(sK, n * 16, kk, 64);
        acc[n] = WMMA_F16(a, b, acc[n]);
      }
    }
#pragma unroll
    for (int n = 0; n < 8; ++n) store_c_f16(sSP, acc[n], wave * 16, n * 16, 128, 0.125f);
  }
  __syncthreads();

  if (tid < 128) {
    // per-row top-16 threshold + masked softmax (in place, f16 row)
    _Float16* row = sSP + tid * 128;
    float top[16];
#pragma unroll
    for (int i = 0; i < 16; ++i) top[i] = -3.4e38f;
    for (int j = 0; j < 128; ++j) {
      float s = (float)row[j];
      if (s > top[15]) {
        int p = 15;
        while (p > 0 && top[p - 1] < s) { top[p] = top[p - 1]; --p; }
        top[p] = s;
      }
    }
    float thr = top[15], mx = top[0];
    float sum = 0.f;
    for (int j = 0; j < 128; ++j) {
      float s = (float)row[j];
      float e = (s >= thr) ? __expf(s - mx) : 0.f;
      sum += e;
      row[j] = (_Float16)e;
    }
    float inv = 1.f / sum;  // sum >= 1 (max term contributes 1)
    for (int j = 0; j < 128; ++j) row[j] = (_Float16)((float)row[j] * inv);
  } else {
    // concurrently stage V^T into sQV: V[v][e] -> sQV[e*128 + v]
    int t = tid - 128;
#pragma unroll
    for (int i = 0; i < 8; ++i) {
      int q = t + i * 128;  // 0..1023 uint4s
      int v = q >> 3, eq = q & 7;
      F16x8 f;
      f.u = *(const uint4*)(vb + base + (size_t)v * 64 + eq * 8);
#pragma unroll
      for (int j = 0; j < 8; ++j) sQV[(eq * 8 + j) * 128 + v] = f.h[j];
    }
  }
  __syncthreads();

  // mixed = P @ V:  M=128, N=64, K=128; B operand from V^T (row-major [e][v])
  {
    v8f acc[4] = {z8, z8, z8, z8};
#pragma unroll
    for (int kk = 0; kk < 128; kk += 32) {
      v16h a = load_a_frag(sSP, wave * 16, kk, 128);
#pragma unroll
      for (int n = 0; n < 4; ++n) {
        v16h b = load_b_frag(sQV, n * 16, kk, 128);
        acc[n] = WMMA_F16(a, b, acc[n]);
      }
    }
#pragma unroll
    for (int n = 0; n < 4; ++n) store_c_f16(sK, acc[n], wave * 16, n * 16, 64, 1.0f);
  }
  __syncthreads();

  // merged-head store: mixed[(bt*128+v)*512 + h*64 + e]
  {
    int v = tid >> 1, half = tid & 1;
    const uint4* s4 = (const uint4*)(sK + v * 64 + half * 32);
    uint4* d4 = (uint4*)(mixed + ((size_t)bt * 128 + v) * 512 + hIdx * 64 + half * 32);
#pragma unroll
    for (int i = 0; i < 4; ++i) d4[i] = s4[i];
  }
}

// ---------------------------------------------------------------------------
// Kernel 3: out = h_val + mixed @ Wo.T, scattered back to [b,V,T,d] fp32.
// grid (1024, 8), block 256.
// ---------------------------------------------------------------------------
__global__ __launch_bounds__(256)
void k3_out_proj(const _Float16* __restrict__ mixed, const float* __restrict__ Wo,
                 const float* __restrict__ h, float* __restrict__ out) {
  __shared__ _Float16 sA[128 * 32];  // 8 KB
  __shared__ _Float16 sB[64 * 32];   // 4 KB
  __shared__ float    sC[128 * 64];  // 32 KB

  const int tid = threadIdx.x;
  const int wave = tid >> 5;
  const int bt = blockIdx.x;       // rows bt*128 .. +127 of mixed (all 128 v)
  const int nB = blockIdx.y * 64;

  const v8f z8 = {0.f, 0.f, 0.f, 0.f, 0.f, 0.f, 0.f, 0.f};
  v8f acc[4] = {z8, z8, z8, z8};

  for (int kt = 0; kt < 16; ++kt) {
    __syncthreads();
#pragma unroll
    for (int i = 0; i < 2; ++i) {
      int q = tid + i * 256;  // 0..511 uint4s (A is already f16)
      int r = q >> 2, c8 = q & 3;
      *(uint4*)(sA + r * 32 + c8 * 8) =
          *(const uint4*)(mixed + ((size_t)bt * 128 + r) * 512 + kt * 32 + c8 * 8);
    }
#pragma unroll
    for (int i = 0; i < 2; ++i) {
      int q = tid + i * 256;  // 0..511 float4s
      int r = q >> 3, c4 = q & 7;
      float4 b4 = *(const float4*)(Wo + (size_t)(nB + r) * 512 + kt * 32 + c4 * 4);
      _Float16* o = sB + r * 32 + c4 * 4;
      o[0] = (_Float16)b4.x; o[1] = (_Float16)b4.y;
      o[2] = (_Float16)b4.z; o[3] = (_Float16)b4.w;
    }
    __syncthreads();
    v16h a = load_a_frag(sA, wave * 16, 0, 32);
#pragma unroll
    for (int n = 0; n < 4; ++n) {
      v16h b = load_b_frag(sB, n * 16, 0, 32);
      acc[n] = WMMA_F16(a, b, acc[n]);
    }
  }
  __syncthreads();
#pragma unroll
  for (int n = 0; n < 4; ++n) store_c_f32(sC, acc[n], wave * 16, n * 16, 64);
  __syncthreads();

  // residual + scatter: row (bt, v) -> out[((b*128+v)*64 + t)*512 + col]
  {
    int r = tid >> 1, half = tid & 1;
    int b = bt >> 6, tt = bt & 63;
    size_t off = (((size_t)b * 128 + r) * 64 + tt) * 512 + nB + half * 32;
    const float* hs = h + off;
    float* os = out + off;
    const float* cs = sC + r * 64 + half * 32;
#pragma unroll
    for (int i = 0; i < 8; ++i) {
      float4 hv = *(const float4*)(hs + i * 4);
      float4 cv = *(const float4*)(cs + i * 4);
      float4 ov;
      ov.x = hv.x + cv.x; ov.y = hv.y + cv.y;
      ov.z = hv.z + cv.z; ov.w = hv.w + cv.w;
      *(float4*)(os + i * 4) = ov;
    }
  }
}

// ---------------------------------------------------------------------------
extern "C" void kernel_launch(void* const* d_in, const int* in_sizes, int n_in,
                              void* d_out, int out_size, void* d_ws, size_t ws_size,
                              hipStream_t stream) {
  (void)in_sizes; (void)n_in; (void)out_size; (void)ws_size;
  const float* h  = (const float*)d_in[0];
  const float* Wq = (const float*)d_in[1];
  const float* Wk = (const float*)d_in[2];
  const float* Wv = (const float*)d_in[3];
  const float* Wo = (const float*)d_in[4];
  float* out = (float*)d_out;

  const size_t NE = (size_t)16 * 128 * 64 * 512;  // 67,108,864 elements
  _Float16* qb = (_Float16*)d_ws;
  _Float16* kb = qb + NE;
  _Float16* vb = kb + NE;
  _Float16* mx = vb + NE;   // merged-head attention output, f16

  k1_qkv_proj<<<dim3(1024, 8, 3), 256, 0, stream>>>(h, Wq, Wk, Wv, qb, kb, vb);
  k2_attention<<<dim3(1024, 8), 256, 0, stream>>>(qb, kb, vb, mx);
  k3_out_proj<<<dim3(1024, 8), 256, 0, stream>>>(mx, Wo, h, out);
}
